// BiMFEBlock_83150566850721
// MI455X (gfx1250) — matrix operand order repaired and checked by reading, not compiled, not run on hardware
//
#include <hip/hip_runtime.h>
#include <hip/hip_bf16.h>

// ---------------- types ----------------
typedef __attribute__((ext_vector_type(16))) __bf16  v16bf;
typedef __attribute__((ext_vector_type(8)))  float   v8f;
typedef __attribute__((ext_vector_type(4)))  unsigned int v4u;
typedef __attribute__((ext_vector_type(4)))  int     v4i;
typedef __attribute__((ext_vector_type(8)))  int     v8i;

union BF16x16 { v16bf v; v4u q[2]; };

static __device__ __forceinline__ v8f vzero8() { v8f z = {}; return z; }

#define HW 16384
#define WID 128

#if defined(__HIP_DEVICE_COMPILE__) && __has_builtin(__builtin_amdgcn_tensor_load_to_lds)
#define USE_TDM 1
#else
#define USE_TDM 0
#endif

#if USE_TDM
// Issue a TDM load of a 2D bf16 tile [tile_m=128 rows x tile_k=32 elts] whose
// rows are strides of `stride_k` elements, into LDS at lds_addr, inserting
// 16B of padding after every 64B row (=> 40-bf16 LDS pitch). OOB rows/cols
// (beyond rem_k/rem_m) are zero-filled by the TDM. One wave issues; EXEC ignored.
static __device__ __forceinline__ void tdm_load_a_tile(unsigned lds_addr,
                                                       const __bf16* gptr,
                                                       int rem_k, int rem_m,
                                                       int stride_k) {
  unsigned long long ga = (unsigned long long)(uintptr_t)gptr;
  v4u g0;
  g0[0] = 1u;                                   // count=1 (valid), user mode
  g0[1] = lds_addr;                             // lds_addr [63:32]
  g0[2] = (unsigned)(ga & 0xffffffffu);         // global_addr lo
  g0[3] = (unsigned)((ga >> 32) & 0x01ffffffu)  // global_addr hi (57-bit)
          | 0x80000000u;                        // type=2 ("image")
  unsigned td0 = (unsigned)rem_k;               // tensor_dim0 (k extent for OOB)
  unsigned td1 = (unsigned)rem_m;               // tensor_dim1 (m extent for OOB)
  v8i g1;
  g1[0] = (int)((1u << 16)      // data_size = 2 bytes
              | (1u << 20)      // pad_enable
              | (3u << 22)      // pad_interval: 16 DWORDs (64B)
              | (3u << 25));    // pad_amount:   4 DWORDs (16B)
  g1[1] = (int)(td0 << 16);                          // [47:32]=atomic_bar=0, [63:48]=td0.lo
  g1[2] = (int)((td0 >> 16) | (td1 << 16));          // [79:64]=td0.hi, [95:80]=td1.lo
  g1[3] = (int)((td1 >> 16) | (32u << 16));          // [111:96]=td1.hi, [127:112]=tile_dim0=32
  g1[4] = (int)(128u);                               // [143:128]=tile_dim1=128, tile_dim2=0
  g1[5] = (int)(unsigned)stride_k;                   // tensor_dim0_stride lo32
  g1[6] = 0;                                         // stride hi16 | tensor_dim1_stride lo
  g1[7] = 0;
  v4i gz4 = {0, 0, 0, 0};
#if defined(__clang_major__) && __clang_major__ >= 23
  v8i gz8 = {0, 0, 0, 0, 0, 0, 0, 0};
  __builtin_amdgcn_tensor_load_to_lds(g0, g1, gz4, gz4, gz8, 0);
#else
  __builtin_amdgcn_tensor_load_to_lds(g0, g1, gz4, gz4, 0);
#endif
}
#endif

// =====================================================================
// K0: convert f32 weights to bf16 (once; makes GEMM A-panels TDM-able)
// =====================================================================
__launch_bounds__(256)
__global__ void cvt_w_kernel(const float* __restrict__ src,
                             __bf16* __restrict__ dst, int n) {
  int i = blockIdx.x * 256 + threadIdx.x;
  if (i < n) dst[i] = (__bf16)src[i];
}

// =====================================================================
// K1: f = x_main + 0.5*downconv(x_prev) + 0.5*pixshuf(upconv(x_next));
//     ln1 = LayerNorm_c(f)  -> bf16.  Block = (b, h, 16-wide w tile).
// =====================================================================
__launch_bounds__(256)
__global__ void fuse_ln1_kernel(const float* __restrict__ xm,
                                const float* __restrict__ xprev,
                                const float* __restrict__ xnext,
                                const float* __restrict__ pad_,
                                const float* __restrict__ pau_,
                                const float* __restrict__ dwn,
                                const float* __restrict__ upw,
                                const float* __restrict__ n1w,
                                const float* __restrict__ n1b,
                                __bf16* __restrict__ ln1) {
  const int w0 = blockIdx.x * 16;
  const int h  = blockIdx.y;
  const int b  = blockIdx.z;
  const int tid = threadIdx.x;

  __shared__ float xp[64 * 3 * 34];   // 64 ci x 3 rows x 34 cols
  __shared__ float xn[256 * 8];       // 256 ci x 8 cols
  __shared__ float ft[128 * 17];      // f tile, padded
  __shared__ float smu[16], sinv[16];

  // stage x_prev tile (stride-2 3x3 halo)
  for (int idx = tid; idx < 64 * 3 * 34; idx += 256) {
    int ci = idx / (3 * 34);
    int rem = idx % (3 * 34);
    int r = rem / 34, col = rem % 34;
    int y = 2 * h - 1 + r, x = 2 * w0 - 1 + col;
    float v = 0.f;
    if (y >= 0 && y < 256 && x >= 0 && x < 256)
      v = xprev[(((size_t)b * 64 + ci) * 256 + y) * 256 + x];
    xp[idx] = v;
  }
  // stage x_next tile
  {
    int y = h >> 1, x0 = w0 >> 1;
    for (int idx = tid; idx < 256 * 8; idx += 256) {
      int ci = idx >> 3, j = idx & 7;
      xn[idx] = xnext[(((size_t)b * 256 + ci) * 64 + y) * 64 + x0 + j];
    }
  }
  __syncthreads();

  const int c = tid & 127, wseg = tid >> 7;   // 8 pixels per thread
  float acd[8], acu[8];
#pragma unroll
  for (int j = 0; j < 8; ++j) { acd[j] = 0.f; acu[j] = 0.f; }

  // down conv (stride 2, pad 1)
  for (int ci = 0; ci < 64; ++ci) {
    float wv[9];
#pragma unroll
    for (int t = 0; t < 9; ++t) wv[t] = dwn[((size_t)c * 64 + ci) * 9 + t];
    const float* xr = &xp[ci * 3 * 34];
#pragma unroll
    for (int r = 0; r < 3; ++r)
#pragma unroll
      for (int kw = 0; kw < 3; ++kw) {
        float ww = wv[r * 3 + kw];
#pragma unroll
        for (int j = 0; j < 8; ++j) {
          int wl = wseg * 8 + j;
          acd[j] = fmaf(ww, xr[r * 34 + 2 * wl + kw], acd[j]);
        }
      }
  }
  // up conv 1x1 + pixel shuffle: channel = 4c + 2*(h&1) + (w&1)
  const int oce = 4 * c + 2 * (h & 1);
  for (int ci = 0; ci < 256; ++ci) {
    float we = upw[(size_t)oce * 256 + ci];
    float wo = upw[(size_t)(oce + 1) * 256 + ci];
    const float* xr = &xn[ci * 8];
#pragma unroll
    for (int j = 0; j < 8; ++j) {
      int wl = wseg * 8 + j;
      acu[j] = fmaf((wl & 1) ? wo : we, xr[wl >> 1], acu[j]);
    }
  }

  const float ad = *pad_, au = *pau_;
  const size_t base = ((size_t)b * 128 + c) * HW + (size_t)h * WID + w0;
#pragma unroll
  for (int j = 0; j < 8; ++j) {
    int wl = wseg * 8 + j;
    float f = xm[base + wl] + ad * acd[j] + au * acu[j];
    ft[c * 17 + wl] = f;
  }
  __syncthreads();
  if (tid < 16) {
    float s = 0.f, s2 = 0.f;
    for (int cc = 0; cc < 128; ++cc) { float v = ft[cc * 17 + tid]; s += v; s2 = fmaf(v, v, s2); }
    float mu = s * (1.f / 128.f);
    float var = s2 * (1.f / 128.f) - mu * mu;
    smu[tid] = mu; sinv[tid] = rsqrtf(var + 1e-5f);
  }
  __syncthreads();
  const float gw = n1w[c], gb = n1b[c];
#pragma unroll
  for (int j = 0; j < 8; ++j) {
    int wl = wseg * 8 + j;
    float v = (ft[c * 17 + wl] - smu[wl]) * sinv[wl] * gw + gb;
    ln1[base + wl] = (__bf16)v;
  }
}

// =====================================================================
// Generic 1x1-conv GEMM: Y[co,p] = sum_ci W[co,ci] * X[ci,p]
// Block tile 128(co) x 64(px), 8 wave32s each 32x32 via 4 WMMA accums.
// A panel DMA'd to LDS by the Tensor Data Mover (wave 0), with TDM row
// padding reproducing the 40-bf16 pitch; B transposed through LDS.
// EPI 0: store bf16.  EPI 1: store f32 of acc + Radd (residual).
// =====================================================================
template <int EPI>
__launch_bounds__(256)
__global__ void gemm1x1_wmma(const __bf16* __restrict__ X,
                             const __bf16* __restrict__ Wb,
                             void* __restrict__ Yv,
                             const float* __restrict__ Radd,
                             int Cin, int Cout) {
  __shared__ __align__(16) __bf16 As[128 * 40];  // [m][k], pitch 40
  __shared__ __align__(16) __bf16 Bt[64 * 40];   // [n][k], pitch 40

  const int n0 = blockIdx.x * 64;
  const int m0 = blockIdx.y * 128;
  const int b  = blockIdx.z;
  const int tid = threadIdx.x;
  const int lane = tid & 31, wid = tid >> 5;
  const int wm = wid >> 1, wn = wid & 1;

  const __bf16* Xb = X + (size_t)b * Cin * HW;
#if USE_TDM
  const unsigned lds_as = (unsigned)(uintptr_t)&As[0];
#endif

  v8f acc[2][2];
#pragma unroll
  for (int i = 0; i < 2; ++i)
#pragma unroll
    for (int j = 0; j < 2; ++j) acc[i][j] = vzero8();

  for (int k0 = 0; k0 < Cin; k0 += 32) {
#if USE_TDM
    // ---- stage A via Tensor Data Mover (wave 0 only; EXEC-independent)
    if (wid == 0) {
      tdm_load_a_tile(lds_as, Wb + (size_t)m0 * Cin + k0,
                      Cin - k0, Cout - m0, Cin);
    }
#else
    // ---- stage A manually (bf16 copy with zero fill)
    {
      int m = tid >> 1, kh = (tid & 1) * 16;
      int gm = m0 + m;
#pragma unroll
      for (int j = 0; j < 16; ++j) {
        int kk = k0 + kh + j;
        As[m * 40 + kh + j] =
            (gm < Cout && kk < Cin) ? Wb[(size_t)gm * Cin + kk] : (__bf16)0.f;
      }
    }
#endif
    // ---- stage B transposed: thread: k=tid/8, 8 contiguous pixels
    {
      int k = tid >> 3, n8 = (tid & 7) * 8;
      int kk = k0 + k;
      if (kk < Cin) {
        v4u raw = *(const v4u*)(Xb + (size_t)kk * HW + n0 + n8);
        const __bf16* e = (const __bf16*)&raw;
#pragma unroll
        for (int j = 0; j < 8; ++j) Bt[(n8 + j) * 40 + k] = e[j];
      } else {
#pragma unroll
        for (int j = 0; j < 8; ++j) Bt[(n8 + j) * 40 + k] = (__bf16)0.f;
      }
    }
    // prefetch next B panel into cache
    if (k0 + 32 < Cin) {
      int kpf = k0 + 32 + (tid >> 3);
      if (kpf < Cin)
        __builtin_prefetch(Xb + (size_t)kpf * HW + n0 + (tid & 7) * 8, 0, 1);
    }
#if USE_TDM
    if (wid == 0) __builtin_amdgcn_s_wait_tensorcnt(0);
#endif
    __syncthreads();

    // ---- operand fetch in WMMA lane layout (ds_load_b128 x2 each)
    const int l15 = lane & 15;
    const int kf8 = 8 * (lane >> 4);
    const int kb  = 16 * (lane >> 4);
    BF16x16 a[2], bo[2];
#pragma unroll
    for (int i = 0; i < 2; ++i) {
      const __bf16* ap = &As[(wm * 32 + i * 16 + l15) * 40];
      a[i].q[0] = *(const v4u*)(ap + kf8);
      a[i].q[1] = *(const v4u*)(ap + 16 + kf8);
    }
#pragma unroll
    for (int j = 0; j < 2; ++j) {
      const __bf16* bp = &Bt[(wn * 32 + j * 16 + l15) * 40 + kf8];
      bo[j].q[0] = *(const v4u*)(bp);
      bo[j].q[1] = *(const v4u*)(bp + 16);
    }
#pragma unroll
    for (int i = 0; i < 2; ++i)
#pragma unroll
      for (int j = 0; j < 2; ++j)
        acc[i][j] = __builtin_amdgcn_wmma_f32_16x16x32_bf16(
            false, a[i].v, false, bo[j].v, (short)0, acc[i][j], false, false);
    __syncthreads();
  }

  // ---- epilogue
  const int l15 = lane & 15;
#pragma unroll
  for (int i = 0; i < 2; ++i)
#pragma unroll
    for (int j = 0; j < 2; ++j) {
      int ng = n0 + wn * 32 + j * 16 + l15;
#pragma unroll
      for (int r = 0; r < 8; ++r) {
        int mg = m0 + wm * 32 + i * 16 + r + 8 * (lane >> 4);
        if (mg < Cout) {
          size_t off = (size_t)b * Cout * HW + (size_t)mg * HW + ng;
          float v = acc[i][j][r];
          if constexpr (EPI == 0) ((__bf16*)Yv)[off] = (__bf16)v;
          else                    ((float*)Yv)[off] = v + Radd[off];
        }
      }
    }
}

// =====================================================================
// K3: depthwise 3x3 (pad 1) over C channels, bf16 in/out
// =====================================================================
__launch_bounds__(128)
__global__ void dw3x3_kernel(const __bf16* __restrict__ in,
                             const float* __restrict__ w9,
                             __bf16* __restrict__ out, int C) {
  const int h = blockIdx.x, ch = blockIdx.y, b = blockIdx.z;
  const int wx = threadIdx.x;
  const __bf16* p = in + ((size_t)b * C + ch) * HW;
  float acc = 0.f;
#pragma unroll
  for (int kh = 0; kh < 3; ++kh) {
    int y = h + kh - 1;
    if ((unsigned)y < 128u) {
#pragma unroll
      for (int kw = 0; kw < 3; ++kw) {
        int x = wx + kw - 1;
        if ((unsigned)x < 128u)
          acc = fmaf(w9[ch * 9 + kh * 3 + kw], (float)p[(size_t)y * WID + x], acc);
      }
    }
  }
  out[((size_t)b * C + ch) * HW + (size_t)h * WID + wx] = (__bf16)acc;
}

// =====================================================================
// K4: inv L2 norms of q,k rows (channels 0..255 of qkvd), over 16384
// =====================================================================
__launch_bounds__(256)
__global__ void rownorm_kernel(const __bf16* __restrict__ qkvd,
                               float* __restrict__ invn) {
  const int c = blockIdx.x;  // 0..255
  const int b = blockIdx.y;
  const __bf16* p = qkvd + ((size_t)b * 384 + c) * HW;
  const int tid = threadIdx.x;
  float s = 0.f;
  for (int i = tid; i < HW; i += 256) { float v = (float)p[i]; s = fmaf(v, v, s); }
  __shared__ float red[256];
  red[tid] = s; __syncthreads();
  for (int st = 128; st > 0; st >>= 1) {
    if (tid < st) red[tid] += red[tid + st];
    __syncthreads();
  }
  if (tid == 0) invn[b * 256 + c] = 1.f / fmaxf(sqrtf(red[0]), 1e-12f);
}

// =====================================================================
// K5: attn logits: per (b,head) 32x32 = q . k^T over K=16384.
// Operands contiguous along K -> direct global b128 loads, no LDS stage.
// 8 waves split K; cross-wave reduce with ds_add_f32.
// =====================================================================
__launch_bounds__(256)
__global__ void attn_logits_kernel(const __bf16* __restrict__ qkvd,
                                   float* __restrict__ logits) {
  const int bh = blockIdx.x;
  const int b = bh >> 2, head = bh & 3;
  const __bf16* q = qkvd + ((size_t)b * 384 + head * 32) * HW;
  const __bf16* k = qkvd + ((size_t)b * 384 + 128 + head * 32) * HW;
  const int tid = threadIdx.x, lane = tid & 31, wid = tid >> 5;
  const int l15 = lane & 15;
  const int kf8 = 8 * (lane >> 4);
  const int kf16 = 16 * (lane >> 4);

  v8f acc[2][2];
#pragma unroll
  for (int i = 0; i < 2; ++i)
#pragma unroll
    for (int j = 0; j < 2; ++j) acc[i][j] = vzero8();

  const int kend = (wid + 1) * 2048;
  for (int kk = wid * 2048; kk < kend; kk += 32) {
    BF16x16 a[2], bo[2];
#pragma unroll
    for (int i = 0; i < 2; ++i) {
      const __bf16* ap = q + (size_t)(l15 + 16 * i) * HW + kk;
      a[i].q[0] = *(const v4u*)(ap + kf8);
      a[i].q[1] = *(const v4u*)(ap + 16 + kf8);
    }
#pragma unroll
    for (int j = 0; j < 2; ++j) {
      const __bf16* kp = k + (size_t)(l15 + 16 * j) * HW + kk + kf8;
      bo[j].q[0] = *(const v4u*)(kp);
      bo[j].q[1] = *(const v4u*)(kp + 16);
    }
    if (kk + 32 < kend) {
      __builtin_prefetch(q + (size_t)l15 * HW + kk + 32, 0, 1);
      __builtin_prefetch(k + (size_t)l15 * HW + kk + 32, 0, 1);
    }
#pragma unroll
    for (int i = 0; i < 2; ++i)
#pragma unroll
      for (int j = 0; j < 2; ++j)
        acc[i][j] = __builtin_amdgcn_wmma_f32_16x16x32_bf16(
            false, a[i].v, false, bo[j].v, (short)0, acc[i][j], false, false);
  }

  __shared__ float sacc[1024];
  for (int i = tid; i < 1024; i += 256) sacc[i] = 0.f;
  __syncthreads();
#pragma unroll
  for (int i = 0; i < 2; ++i)
#pragma unroll
    for (int j = 0; j < 2; ++j) {
      int n = j * 16 + l15;
#pragma unroll
      for (int r = 0; r < 8; ++r) {
        int m = i * 16 + r + 8 * (lane >> 4);
        atomicAdd(&sacc[m * 32 + n], acc[i][j][r]);   // ds_add_f32
      }
    }
  __syncthreads();
  for (int i = tid; i < 1024; i += 256) logits[(size_t)bh * 1024 + i] = sacc[i];
}

// =====================================================================
// K6: softmax over d with folded q/k norms and temperature -> bf16
// =====================================================================
__launch_bounds__(32)
__global__ void softmax32_kernel(const float* __restrict__ logits,
                                 const float* __restrict__ invn,
                                 const float* __restrict__ temp,
                                 __bf16* __restrict__ attnw) {
  const int bh = blockIdx.x;
  const int b = bh >> 2, head = bh & 3;
  const int c = threadIdx.x;
  const float iq = invn[b * 256 + head * 32 + c];
  const float T = temp[head];
  float row[32];
  float mx = -3.4e38f;
#pragma unroll
  for (int d = 0; d < 32; ++d) {
    float ik = invn[b * 256 + 128 + head * 32 + d];
    row[d] = logits[(size_t)bh * 1024 + c * 32 + d] * iq * ik * T;
    mx = fmaxf(mx, row[d]);
  }
  float s = 0.f;
#pragma unroll
  for (int d = 0; d < 32; ++d) { row[d] = __expf(row[d] - mx); s += row[d]; }
  const float r = 1.f / s;
#pragma unroll
  for (int d = 0; d < 32; ++d)
    attnw[(size_t)bh * 1024 + c * 32 + d] = (__bf16)(row[d] * r);
}

// =====================================================================
// K7: out = attnw(32x32) @ v(32x16384) per (b,head). K=32 -> 1 WMMA step.
// v tile transposed through LDS; A loaded direct (K contiguous).
// =====================================================================
__launch_bounds__(256)
__global__ void attn_apply_kernel(const __bf16* __restrict__ attnw,
                                  const __bf16* __restrict__ qkvd,
                                  __bf16* __restrict__ attnv) {
  const int n0 = blockIdx.x * 256;
  const int bh = blockIdx.y;
  const int b = bh >> 2, head = bh & 3;
  const __bf16* v = qkvd + ((size_t)b * 384 + 256 + head * 32) * HW;
  const __bf16* aw = attnw + (size_t)bh * 1024;

  __shared__ __align__(16) __bf16 vt[256 * 40];  // [n][k]
  const int tid = threadIdx.x, lane = tid & 31, wid = tid >> 5;

  {  // stage v transposed: thread: d=tid/8, 32 pixels
    int d = tid >> 3, nb = (tid & 7) * 32;
    const __bf16* src = v + (size_t)d * HW + n0 + nb;
    v4u raw[4];
#pragma unroll
    for (int t = 0; t < 4; ++t) raw[t] = *(const v4u*)(src + t * 8);
    const __bf16* e = (const __bf16*)raw;
#pragma unroll
    for (int j = 0; j < 32; ++j) vt[(nb + j) * 40 + d] = e[j];
  }
  __syncthreads();

  const int l15 = lane & 15;
  const int kf8 = 8 * (lane >> 4);
  const int kb  = 16 * (lane >> 4);
  BF16x16 a[2], bo[2];
#pragma unroll
  for (int i = 0; i < 2; ++i) {
    const __bf16* ap = aw + (size_t)(l15 + 16 * i) * 32;
    a[i].q[0] = *(const v4u*)(ap + kf8);
    a[i].q[1] = *(const v4u*)(ap + 16 + kf8);
  }
#pragma unroll
  for (int j = 0; j < 2; ++j) {
    const __bf16* bp = &vt[(wid * 32 + j * 16 + l15) * 40 + kf8];
    bo[j].q[0] = *(const v4u*)(bp);
    bo[j].q[1] = *(const v4u*)(bp + 16);
  }
  v8f acc[2][2];
#pragma unroll
  for (int i = 0; i < 2; ++i)
#pragma unroll
    for (int j = 0; j < 2; ++j) {
      acc[i][j] = vzero8();
      acc[i][j] = __builtin_amdgcn_wmma_f32_16x16x32_bf16(
          false, a[i].v, false, bo[j].v, (short)0, acc[i][j], false, false);
    }
#pragma unroll
  for (int i = 0; i < 2; ++i)
#pragma unroll
    for (int j = 0; j < 2; ++j) {
      int ng = n0 + wid * 32 + j * 16 + l15;
#pragma unroll
      for (int r = 0; r < 8; ++r) {
        int mg = i * 16 + r + 8 * (lane >> 4);
        attnv[((size_t)b * 128 + head * 32 + mg) * HW + ng] = (__bf16)acc[i][j][r];
      }
    }
}

// =====================================================================
// K9: channel LayerNorm of x1 (f32) -> bf16
// =====================================================================
__launch_bounds__(128)
__global__ void ln_c_kernel(const float* __restrict__ x,
                            const float* __restrict__ gw,
                            const float* __restrict__ gb,
                            __bf16* __restrict__ out) {
  const int h = blockIdx.x, b = blockIdx.y;
  const int wx = threadIdx.x;
  const size_t base = ((size_t)b * 128) * HW + (size_t)h * WID + wx;
  float s = 0.f, s2 = 0.f;
  for (int c = 0; c < 128; ++c) {
    float v = x[base + (size_t)c * HW];
    s += v; s2 = fmaf(v, v, s2);
  }
  const float mu = s * (1.f / 128.f);
  const float inv = rsqrtf(s2 * (1.f / 128.f) - mu * mu + 1e-5f);
  for (int c = 0; c < 128; ++c) {
    float v = (x[base + (size_t)c * HW] - mu) * inv * gw[c] + gb[c];
    out[base + (size_t)c * HW] = (__bf16)v;
  }
}

// =====================================================================
// K11: fused depthwise 3x3 on both FFN halves + exact-GELU gate -> bf16
// =====================================================================
__launch_bounds__(128)
__global__ void ffn_dw_gate_kernel(const __bf16* __restrict__ g,
                                   const float* __restrict__ w9,
                                   __bf16* __restrict__ hout) {
  const int h = blockIdx.x, c = blockIdx.y, b = blockIdx.z;
  const int wx = threadIdx.x;
  const __bf16* p1 = g + ((size_t)b * 680 + c) * HW;
  const __bf16* p2 = g + ((size_t)b * 680 + c + 340) * HW;
  float a1 = 0.f, a2 = 0.f;
#pragma unroll
  for (int kh = 0; kh < 3; ++kh) {
    int y = h + kh - 1;
    if ((unsigned)y < 128u) {
#pragma unroll
      for (int kw = 0; kw < 3; ++kw) {
        int x = wx + kw - 1;
        if ((unsigned)x < 128u) {
          int t = kh * 3 + kw;
          a1 = fmaf(w9[c * 9 + t],         (float)p1[(size_t)y * WID + x], a1);
          a2 = fmaf(w9[(c + 340) * 9 + t], (float)p2[(size_t)y * WID + x], a2);
        }
      }
    }
  }
  float ge = 0.5f * a1 * (1.f + erff(a1 * 0.70710678118654752f));
  hout[((size_t)b * 340 + c) * HW + (size_t)h * WID + wx] = (__bf16)(ge * a2);
}

// =====================================================================
// launcher
// =====================================================================
extern "C" void kernel_launch(void* const* d_in, const int* in_sizes, int n_in,
                              void* d_out, int out_size, void* d_ws, size_t ws_size,
                              hipStream_t stream) {
  const float* x_main   = (const float*)d_in[0];
  const float* x_prev   = (const float*)d_in[1];
  const float* x_next   = (const float*)d_in[2];
  const float* a_down   = (const float*)d_in[3];
  const float* a_up     = (const float*)d_in[4];
  const float* down_w   = (const float*)d_in[5];
  const float* up_w     = (const float*)d_in[6];
  const float* n1w      = (const float*)d_in[7];
  const float* n1b      = (const float*)d_in[8];
  const float* temp     = (const float*)d_in[9];
  const float* qkv_w    = (const float*)d_in[10];
  const float* qkv_dw_w = (const float*)d_in[11];
  const float* proj_w   = (const float*)d_in[12];
  const float* n2w      = (const float*)d_in[13];
  const float* n2b      = (const float*)d_in[14];
  const float* ffn_in_w = (const float*)d_in[15];
  const float* ffn_dw_w = (const float*)d_in[16];
  const float* ffn_out_w= (const float*)d_in[17];

  // workspace layout (bytes); ln2 aliases ln1, h aliases qkv (dead by then)
  const size_t SZ_LN  = (size_t)8 * 128 * HW * 2;   //  32 MB
  const size_t SZ_QKV = (size_t)8 * 384 * HW * 2;   //  96 MB
  const size_t SZ_X1  = (size_t)8 * 128 * HW * 4;   //  64 MB
  const size_t SZ_G   = (size_t)8 * 680 * HW * 2;   // 170 MB
  char* ws = (char*)d_ws;
  size_t off = 0;
  __bf16* ln1   = (__bf16*)(ws + off);               // also ln2
  size_t off_ln = off; off += SZ_LN;
  __bf16* qkv   = (__bf16*)(ws + off);               // also h (89MB <= 96MB)
  size_t off_qkv = off; off += SZ_QKV;
  __bf16* qkvd  = (__bf16*)(ws + off); off += SZ_QKV;
  __bf16* attnv = (__bf16*)(ws + off); off += SZ_LN;
  float*  x1    = (float*)(ws + off);  off += SZ_X1;
  __bf16* g     = (__bf16*)(ws + off); off += SZ_G;
  float*  invn  = (float*)(ws + off);  off += 8 * 256 * 4;
  float*  logit = (float*)(ws + off);  off += 8 * 4 * 32 * 32 * 4;
  __bf16* attnw = (__bf16*)(ws + off); off += 8 * 4 * 32 * 32 * 2;
  // bf16 weight copies (TDM-able A panels)
  __bf16* wb_qkv  = (__bf16*)(ws + off); off += (size_t)384 * 128 * 2;
  __bf16* wb_proj = (__bf16*)(ws + off); off += (size_t)128 * 128 * 2;
  __bf16* wb_fin  = (__bf16*)(ws + off); off += (size_t)680 * 128 * 2;
  __bf16* wb_fout = (__bf16*)(ws + off); off += (size_t)128 * 340 * 2;
  __bf16* ln2   = (__bf16*)(ws + off_ln);
  __bf16* hbuf  = (__bf16*)(ws + off_qkv);
  (void)ws_size; (void)n_in; (void)in_sizes; (void)out_size;

  // 0) one-time f32->bf16 weight conversion
  cvt_w_kernel<<<(384 * 128 + 255) / 256, 256, 0, stream>>>(qkv_w, wb_qkv, 384 * 128);
  cvt_w_kernel<<<(128 * 128 + 255) / 256, 256, 0, stream>>>(proj_w, wb_proj, 128 * 128);
  cvt_w_kernel<<<(680 * 128 + 255) / 256, 256, 0, stream>>>(ffn_in_w, wb_fin, 680 * 128);
  cvt_w_kernel<<<(128 * 340 + 255) / 256, 256, 0, stream>>>(ffn_out_w, wb_fout, 128 * 340);

  // 1) input fusion + LN1
  fuse_ln1_kernel<<<dim3(8, 128, 8), 256, 0, stream>>>(
      x_main, x_prev, x_next, a_down, a_up, down_w, up_w, n1w, n1b, ln1);
  // 2) qkv 1x1 GEMM (384x128)
  gemm1x1_wmma<0><<<dim3(256, 3, 8), 256, 0, stream>>>(ln1, wb_qkv, qkv, nullptr, 128, 384);
  // 3) depthwise 3x3 on qkv
  dw3x3_kernel<<<dim3(128, 384, 8), 128, 0, stream>>>(qkv, qkv_dw_w, qkvd, 384);
  // 4) q/k row inverse norms
  rownorm_kernel<<<dim3(256, 8), 256, 0, stream>>>(qkvd, invn);
  // 5) attention logits (32x32, K=16384) via WMMA, ds_add reduce
  attn_logits_kernel<<<dim3(32), 256, 0, stream>>>(qkvd, logit);
  // 6) softmax with folded norms + temperature
  softmax32_kernel<<<dim3(32), 32, 0, stream>>>(logit, invn, temp, attnw);
  // 7) attn @ v
  attn_apply_kernel<<<dim3(64, 32), 256, 0, stream>>>(attnw, qkvd, attnv);
  // 8) proj GEMM + residual(x_main) -> x1 (f32)
  gemm1x1_wmma<1><<<dim3(256, 1, 8), 256, 0, stream>>>(attnv, wb_proj, x1, x_main, 128, 128);
  // 9) LN2
  ln_c_kernel<<<dim3(128, 8), 128, 0, stream>>>(x1, n2w, n2b, ln2);
  // 10) FFN-in GEMM (680x128)
  gemm1x1_wmma<0><<<dim3(256, 6, 8), 256, 0, stream>>>(ln2, wb_fin, g, nullptr, 128, 680);
  // 11) fused depthwise + GELU gate -> h (340 ch)
  ffn_dw_gate_kernel<<<dim3(128, 340, 8), 128, 0, stream>>>(g, ffn_dw_w, hbuf);
  // 12) FFN-out GEMM (128x340) + residual(x1) -> d_out (f32)
  gemm1x1_wmma<1><<<dim3(256, 1, 8), 256, 0, stream>>>(hbuf, wb_fout, (float*)d_out, x1, 340, 128);
}
